// AdjAttenAgger_89223650607749
// MI455X (gfx1250) — matrix operand order repaired and checked by reading, not compiled, not run on hardware
//
#include <hip/hip_runtime.h>
#include <hip/hip_bf16.h>
#include <math.h>

// ---------------- problem constants (match reference) ----------------
#define NQ   16384
#define NK   4096
#define QDIM 512
#define KDIM 256
#define MID  256

typedef __bf16 bf16_t;
typedef __attribute__((ext_vector_type(16))) __bf16 v16bf;
typedef __attribute__((ext_vector_type(8)))  __bf16 v8bf;
typedef __attribute__((ext_vector_type(8)))  float  v8f;
typedef __attribute__((ext_vector_type(4)))  unsigned int u32x4;
typedef __attribute__((ext_vector_type(8)))  int          i32x8;
typedef __attribute__((ext_vector_type(4)))  int          i32x4;

#if defined(__has_builtin)
#if __has_builtin(__builtin_amdgcn_tensor_load_to_lds)
#define HAS_TDM 1
#endif
#endif
#ifndef HAS_TDM
#define HAS_TDM 0
#endif

// =====================================================================
// Kernel 1: Y[rows, MID](bf16) = X[rows, kdim](f32) @ W[MID, kdim]^T + b
// One 16x16 C tile per (wave, t); K consumed 32 at a time via
// v_wmma_f32_16x16x32_bf16.  Block = 256 thr = 8 waves covers 16 rows
// x 256 cols (each wave does 2 column tiles).
// =====================================================================
__global__ __launch_bounds__(256) void proj_gemm_bf16(
    const float* __restrict__ X, const float* __restrict__ W,
    const float* __restrict__ bias, bf16_t* __restrict__ Y, int kdim)
{
  const int lane  = threadIdx.x & 31;
  const int wave  = threadIdx.x >> 5;
  const int rbase = blockIdx.x * 16;
  const int m     = lane & 15;          // A row / B col within tile
  const int hi    = lane >> 4;          // half-wave select
  const int nchunk = kdim >> 5;

  for (int t = 0; t < 2; ++t) {
    const int col0 = (wave * 2 + t) * 16;
    v8f acc = {};
    for (int c = 0; c < nchunk; ++c) {
      // A tile 16x32: lane holds row m, K = c*32 + hi*8 + {0..7, 16..23}
      const float* ap = X + (size_t)(rbase + m) * kdim + c * 32 + hi * 8;
      v16bf a;
      v8bf alo, ahi;
#pragma unroll
      for (int e = 0; e < 8; ++e) { alo[e] = (__bf16)ap[e]; ahi[e] = (__bf16)ap[e + 16]; }
      *((v8bf*)&a)     = alo;
      *((v8bf*)&a + 1) = ahi;
      // B tile 32x16: lane holds col (col0+m), K = c*32 + hi*8 + {0..7, 16..23}
      const float* bp = W + (size_t)(col0 + m) * kdim + c * 32 + hi * 8;
      v16bf b;
#pragma unroll
      for (int e = 0; e < 16; ++e) b[e] = (__bf16)bp[e < 8 ? e : e + 8];
      acc = __builtin_amdgcn_wmma_f32_16x16x32_bf16(false, a, false, b,
                                                    (short)0, acc, false, false);
    }
    // C layout: VGPR r -> row (r + 8*hi), col = col0 + m
#pragma unroll
    for (int r = 0; r < 8; ++r) {
      const int row = rbase + r + 8 * hi;
      const int col = col0 + m;
      Y[(size_t)row * MID + col] = (bf16_t)(acc[r] + bias[col]);
    }
  }
}

// =====================================================================
// Kernel 2: VT[c, r](bf16) = w[r] * S[r, c]   (transposed so PV B-tiles
// are contiguous 32B loads)
// =====================================================================
__global__ __launch_bounds__(256) void scale_transpose_v(
    const float* __restrict__ S, const float* __restrict__ w,
    bf16_t* __restrict__ VT)
{
  const int idx = blockIdx.x * 256 + threadIdx.x;   // column-major walk
  const int c = idx / NK;
  const int r = idx - c * NK;
  VT[(size_t)c * NK + r] = (bf16_t)(w[r] * S[(size_t)r * KDIM + c]);
}

// ---------------- TDM: load a [32 x MID] bf16 tile into LDS ----------------
#if HAS_TDM
__device__ __forceinline__ void tdm_load_ktile(const bf16_t* gsrc,
                                               unsigned lds_byte_addr)
{
  const unsigned long long ga = (unsigned long long)(const void*)gsrc;
  u32x4 g0;
  g0[0] = 1u;                                      // count=1, user descriptor
  g0[1] = lds_byte_addr;                           // lds_addr[63:32]
  g0[2] = (unsigned)(ga & 0xFFFFFFFFu);            // global_addr lo
  g0[3] = (unsigned)((ga >> 32) & 0x01FFFFFFu)     // global_addr hi
        | (2u << 30);                              // type = 2 (image)
  i32x8 g1;
  g1[0] = (1 << 16);                               // data_size = 1 (2 bytes)
  g1[1] = (int)((unsigned)MID << 16);              // tensor_dim0 = MID
  g1[2] = (int)((unsigned)NK  << 16);              // tensor_dim1 = NK
  g1[3] = (int)((unsigned)MID << 16);              // tile_dim0 = MID
  g1[4] = 32;                                      // tile_dim1 = 32 rows
  g1[5] = MID;                                     // tensor_dim0_stride
  g1[6] = 0;
  g1[7] = 0;
  i32x4 z4 = {0, 0, 0, 0};
  i32x8 z8 = {0, 0, 0, 0, 0, 0, 0, 0};
  // clang-23 / therock-10.0 signature: (g0, g1, g2, g3, g4, cpol)
  __builtin_amdgcn_tensor_load_to_lds(g0, g1, z4, z4, z8, 0);
}
#endif

// =====================================================================
// Kernel 3: flash attention.  4 waves/block, 16 Q rows per wave.
// Qb held in registers, K tiles TDM double-buffered through LDS,
// P re-layout (C->A) staged through LDS, V read from resident-in-L2 VT.
// =====================================================================
#define FA_WAVES 4
__global__ __launch_bounds__(128) void flash_attn(
    const bf16_t* __restrict__ Qb, const bf16_t* __restrict__ Kb,
    const bf16_t* __restrict__ VT, const int* __restrict__ mask,
    float* __restrict__ Out)
{
  __shared__ bf16_t kTile[2][32 * MID];            // 2 x 16 KB double buffer
  __shared__ bf16_t pTile[FA_WAVES][16 * 32];      // per-wave P staging

  const int lane  = threadIdx.x & 31;
  const int wave  = threadIdx.x >> 5;
  const int qbase = blockIdx.x * (FA_WAVES * 16) + wave * 16;
  const int m     = lane & 15;
  const int hi    = lane >> 4;

  // ---- load Q A-tiles (16 x 256 bf16 = 8 chunks of 16x32) into VGPRs ----
  v16bf aQ[8];
  {
    const bf16_t* qrow = Qb + (size_t)(qbase + m) * MID;
#pragma unroll
    for (int c = 0; c < 8; ++c) {
      const bf16_t* p = qrow + c * 32 + hi * 8;
      *((v8bf*)&aQ[c])     = *(const v8bf*)p;
      *((v8bf*)&aQ[c] + 1) = *(const v8bf*)(p + 16);
    }
  }

  v8f o[16];
#pragma unroll
  for (int t = 0; t < 16; ++t) o[t] = (v8f){};
  float mst[8], lst[8];
#pragma unroll
  for (int r = 0; r < 8; ++r) { mst[r] = -3.0e38f; lst[r] = 0.0f; }

#if HAS_TDM
  if (wave == 0) tdm_load_ktile(Kb, (unsigned)(size_t)&kTile[0][0]);
#endif

  for (int j = 0; j < NK; j += 32) {
    const int cur = (j >> 5) & 1;
#if HAS_TDM
    if (wave == 0) {
      if (j + 32 < NK) {
        tdm_load_ktile(Kb + (size_t)(j + 32) * MID,
                       (unsigned)(size_t)&kTile[cur ^ 1][0]);
        __builtin_amdgcn_s_wait_tensorcnt(1);      // current tile landed
      } else {
        __builtin_amdgcn_s_wait_tensorcnt(0);
      }
    }
    __syncthreads();
#else
    __syncthreads();
    for (int idx = threadIdx.x; idx < (32 * MID) / 8; idx += blockDim.x)
      ((v8bf*)&kTile[cur][0])[idx] =
          ((const v8bf*)(Kb + (size_t)j * MID))[idx];
    __syncthreads();
#endif

    // ---- scores: two 16x16 tiles over this 32-column window ----
    v8f s0 = {}, s1 = {};
#pragma unroll
    for (int c = 0; c < 8; ++c) {
      // B[k][n] = K[j + sub*16 + n][c*32 + hi*8 + {0..7,16..23}]
      const bf16_t* kp0 = &kTile[cur][(size_t)m * MID + c * 32 + hi * 8];
      const bf16_t* kp1 = &kTile[cur][(size_t)(16 + m) * MID + c * 32 + hi * 8];
      v16bf b0, b1;
      *((v8bf*)&b0)     = *(const v8bf*)kp0;
      *((v8bf*)&b0 + 1) = *(const v8bf*)(kp0 + 16);
      *((v8bf*)&b1)     = *(const v8bf*)kp1;
      *((v8bf*)&b1 + 1) = *(const v8bf*)(kp1 + 16);
      s0 = __builtin_amdgcn_wmma_f32_16x16x32_bf16(false, aQ[c], false, b0,
                                                   (short)0, s0, false, false);
      s1 = __builtin_amdgcn_wmma_f32_16x16x32_bf16(false, aQ[c], false, b1,
                                                   (short)0, s1, false, false);
    }

    // ---- mask + online softmax (C layout: row = r + 8*hi, col = m) ----
    const float scale = 0.0625f;                   // 1/sqrt(MID)
#pragma unroll
    for (int r = 0; r < 8; ++r) {
      const int row = qbase + r + 8 * hi;
      const int* mrow = mask + (size_t)row * NK + j;
      __builtin_prefetch(mrow + 32, 0, 1);         // stream the mask ahead
      float x0 = (mrow[m]      != 0) ? s0[r] * scale : -3.0e38f;
      float x1 = (mrow[16 + m] != 0) ? s1[r] * scale : -3.0e38f;
      float rmax = fmaxf(x0, x1);
#pragma unroll
      for (int off = 8; off >= 1; off >>= 1)       // reduce within 16-lane half
        rmax = fmaxf(rmax, __shfl_xor(rmax, off, 32));
      const float mnew  = fmaxf(mst[r], rmax);
      const float alpha = __expf(mst[r] - mnew);
      const float p0 = __expf(x0 - mnew);
      const float p1 = __expf(x1 - mnew);
      float rsum = p0 + p1;
#pragma unroll
      for (int off = 8; off >= 1; off >>= 1)
        rsum += __shfl_xor(rsum, off, 32);
      lst[r] = lst[r] * alpha + rsum;
      mst[r] = mnew;
#pragma unroll
      for (int t = 0; t < 16; ++t) o[t][r] *= alpha;   // rescale accumulators
      const int prow = r + 8 * hi;
      pTile[wave][prow * 32 + m]      = (bf16_t)p0;
      pTile[wave][prow * 32 + 16 + m] = (bf16_t)p1;
    }
    __syncthreads();

    // ---- re-load P in A layout: lane m = row, K = hi*8 + {0..7,16..23} ----
    v16bf pA;
    {
      const bf16_t* pp = &pTile[wave][m * 32 + hi * 8];
      *((v8bf*)&pA)     = *(const v8bf*)pp;
      *((v8bf*)&pA + 1) = *(const v8bf*)(pp + 16);
    }

    // ---- PV: B[k][n] = VT[(t*16+n)*NK + j + hi*8 + {0..7,16..23}] ----
#pragma unroll
    for (int t = 0; t < 16; ++t) {
      const bf16_t* vp = VT + (size_t)(t * 16 + m) * NK + j + hi * 8;
      v16bf bV;
      *((v8bf*)&bV)     = *(const v8bf*)vp;
      *((v8bf*)&bV + 1) = *(const v8bf*)(vp + 16);
      o[t] = __builtin_amdgcn_wmma_f32_16x16x32_bf16(false, pA, false, bV,
                                                     (short)0, o[t], false, false);
    }
    __syncthreads();   // protect kTile[cur^1] before next TDM overwrite
  }

  // ---- epilogue: normalize and store f32 ----
#pragma unroll
  for (int r = 0; r < 8; ++r) {
    const float inv = 1.0f / fmaxf(lst[r], 1e-30f);
    const int row = qbase + r + 8 * hi;
#pragma unroll
    for (int t = 0; t < 16; ++t)
      Out[(size_t)row * KDIM + t * 16 + m] = o[t][r] * inv;
  }
}

// =====================================================================
// Host-side launcher
// =====================================================================
extern "C" void kernel_launch(void* const* d_in, const int* in_sizes, int n_in,
                              void* d_out, int out_size, void* d_ws, size_t ws_size,
                              hipStream_t stream) {
  const float* G    = (const float*)d_in[0];   // [NQ, QDIM]
  const float* S    = (const float*)d_in[1];   // [NK, KDIM]
  const float* w    = (const float*)d_in[2];   // [NK]
  const int*   mask = (const int*)  d_in[3];   // [NQ, NK]
  const float* Wq   = (const float*)d_in[4];   // [MID, QDIM]
  const float* bq   = (const float*)d_in[5];   // [MID]
  const float* Wk   = (const float*)d_in[6];   // [MID, KDIM]
  const float* bk   = (const float*)d_in[7];   // [MID]
  float* out = (float*)d_out;                  // [NQ, KDIM] f32

  // workspace layout: Qb (8MB) | Kb (2MB) | VT (2MB)
  bf16_t* Qb = (bf16_t*)d_ws;
  bf16_t* Kb = Qb + (size_t)NQ * MID;
  bf16_t* VT = Kb + (size_t)NK * MID;

  proj_gemm_bf16<<<NQ / 16, 256, 0, stream>>>(G, Wq, bq, Qb, QDIM);
  proj_gemm_bf16<<<NK / 16, 256, 0, stream>>>(S, Wk, bk, Kb, KDIM);
  scale_transpose_v<<<(NK * KDIM) / 256, 256, 0, stream>>>(S, w, VT);
  flash_attn<<<NQ / (FA_WAVES * 16), FA_WAVES * 32, 0, stream>>>(Qb, Kb, VT, mask, out);
}
